// WindowMultiHeadAttention_1803886264576
// MI455X (gfx1250) — matrix operand order repaired and checked by reading, not compiled, not run on hardware
//
#include <hip/hip_runtime.h>
#include <hip/hip_bf16.h>

typedef _Float16 f16;
typedef __attribute__((ext_vector_type(4)))  _Float16 v4h;
typedef __attribute__((ext_vector_type(8)))  _Float16 v8h;
typedef __attribute__((ext_vector_type(16))) _Float16 v16h;
typedef __attribute__((ext_vector_type(8)))  float    v8f;

#define DIM     128
#define HEADS   4
#define HD      32
#define L_TOK   256
#define BW      1024
#define NWIN    64
#define MHID    384

// padded LDS row strides (halfs); all multiples of 8 halfs (16B) and chosen so
// row-stride in banks is coprime-ish with 64 banks (no 16-lane conflicts)
#define QSTR   40     // 32-wide rows  (80B  -> 20 words/row)
#define GSTR   136    // 128-wide rows (272B -> 68 words/row)
#define VSTR   264    // 256-wide rows (528B -> 132 words/row)

static __device__ __forceinline__ v8f wmma16(v16h a, v16h b, v8f c) {
  return __builtin_amdgcn_wmma_f32_16x16x32_f16(false, a, false, b, (short)0, c,
                                                false, false);
}
static __device__ __forceinline__ v8f zero8() {
  v8f z;
#pragma unroll
  for (int i = 0; i < 8; ++i) z[i] = 0.0f;
  return z;
}
static __device__ __forceinline__ v16h cat8(v8h lo, v8h hi) {
  return __builtin_shufflevector(lo, hi, 0, 1, 2, 3, 4, 5, 6, 7, 8, 9, 10, 11,
                                 12, 13, 14, 15);
}
// A-operand fragment: two contiguous 8-half runs at K = hb*8 and 16+hb*8
static __device__ __forceinline__ v16h fragA(const f16* row, int hb) {
  v8h lo = *(const v8h*)(row + hb * 8);
  v8h hi = *(const v8h*)(row + 16 + hb * 8);
  return cat8(lo, hi);
}
// B-operand fragment: two contiguous 8-half runs at K = hb*8 and 16+hb*8
static __device__ __forceinline__ v16h fragB(const f16* row, int hb) {
  v8h lo = *(const v8h*)(row + hb * 8);
  v8h hi = *(const v8h*)(row + 16 + hb * 8);
  return cat8(lo, hi);
}

// ---------------------------------------------------------------------------
// Kernel 0: convert qkv_w (384x128) and proj_w (128x128) weights to f16
// ---------------------------------------------------------------------------
__global__ void convert_w_kernel(const float* __restrict__ qkv_w,
                                 const float* __restrict__ proj_w,
                                 f16* __restrict__ qkvw16,
                                 f16* __restrict__ projw16) {
  int i = blockIdx.x * 256 + threadIdx.x;  // grid = 192 blocks -> 49152
  if (i < MHID * DIM) qkvw16[i] = (f16)qkv_w[i];
  if (i < DIM * DIM)  projw16[i] = (f16)proj_w[i];
}

// ---------------------------------------------------------------------------
// Kernel 1: continuous relative position bias via meta MLP -> (H, L, L) f32
// ---------------------------------------------------------------------------
__global__ __launch_bounds__(256) void bias_kernel(
    const float* __restrict__ w1, const float* __restrict__ b1,
    const float* __restrict__ w2, const float* __restrict__ b2,
    float* __restrict__ biasws) {
  const int i = blockIdx.x;   // q position
  const int j = threadIdx.x;  // k position
  float dy = (float)((i >> 4) - (j >> 4));
  float dx = (float)((i & 15) - (j & 15));
  float ry = (dy >= 0.0f ? 1.0f : -1.0f) * log1pf(fabsf(dy));
  float rx = (dx >= 0.0f ? 1.0f : -1.0f) * log1pf(fabsf(dx));
  float acc0 = 0.f, acc1 = 0.f, acc2 = 0.f, acc3 = 0.f;
  for (int t = 0; t < MHID; ++t) {
    float hval = fmaxf(ry * w1[t * 2 + 0] + rx * w1[t * 2 + 1] + b1[t], 0.0f);
    acc0 += hval * w2[0 * MHID + t];
    acc1 += hval * w2[1 * MHID + t];
    acc2 += hval * w2[2 * MHID + t];
    acc3 += hval * w2[3 * MHID + t];
  }
  int p = i * L_TOK + j;
  biasws[0 * 65536 + p] = acc0 + b2[0];
  biasws[1 * 65536 + p] = acc1 + b2[1];
  biasws[2 * 65536 + p] = acc2 + b2[2];
  biasws[3 * 65536 + p] = acc3 + b2[3];
}

// ---------------------------------------------------------------------------
// Kernel 2: QKV GEMM.  out(262144 x 384) = x(262144 x 128) @ qkv_w^T + b
// A staged [row][k] f16; B staged row-major [n][k] (K-contiguous -> b128
// fragment loads).  8 waves, wave tile 32x64, 8 WMMA accumulators.
// ---------------------------------------------------------------------------
__global__ __launch_bounds__(256) void qkv_gemm_kernel(
    const float* __restrict__ x, const f16* __restrict__ w16,
    const float* __restrict__ qkv_b, f16* __restrict__ qkv16) {
  const int mb = blockIdx.x, nb = blockIdx.y;
  __shared__ __align__(16) f16 As[128 * GSTR];
  __shared__ __align__(16) f16 Bs[128 * GSTR];
  const int tid = threadIdx.x;

  // stage x tile (f32 -> f16), 8-byte packed stores
#pragma unroll
  for (int c = 0; c < 16; ++c) {
    int g = c * 256 + tid;  // 4096 float4 groups
    int row = g >> 5, col4 = (g & 31) << 2;
    const float4 v =
        ((const float4*)x)[(size_t)(mb * 128 + row) * 32 + (g & 31)];
    v4h hv;
    hv[0] = (f16)v.x; hv[1] = (f16)v.y; hv[2] = (f16)v.z; hv[3] = (f16)v.w;
    *(v4h*)&As[row * GSTR + col4] = hv;
  }
  // stage weight tile row-major: Bs[n][k] = w16[(nb*128+n)*128 + k]
#pragma unroll
  for (int c = 0; c < 8; ++c) {
    int g = c * 256 + tid;  // 2048 groups of 8 halfs
    int n = g >> 4, k8 = (g & 15) << 3;
    uint4 raw = *(const uint4*)&w16[(size_t)(nb * 128 + n) * 128 + k8];
    *(uint4*)&Bs[n * GSTR + k8] = raw;
  }
  __syncthreads();

  const int lane = tid & 31, wid = tid >> 5;
  const int lnk = lane & 15, hb = lane >> 4;
  const int Mb = (wid & 3) * 32, Nb = (wid >> 2) * 64;

  v8f acc[2][4];
#pragma unroll
  for (int mi = 0; mi < 2; ++mi)
#pragma unroll
    for (int ni = 0; ni < 4; ++ni) acc[mi][ni] = zero8();

#pragma unroll
  for (int kk = 0; kk < 128; kk += 32) {
    v16h a[2];
#pragma unroll
    for (int mi = 0; mi < 2; ++mi)
      a[mi] = fragA(&As[(Mb + mi * 16 + lnk) * GSTR + kk], hb);
    v16h bf[4];
#pragma unroll
    for (int ni = 0; ni < 4; ++ni)
      bf[ni] = fragB(&Bs[(Nb + ni * 16 + lnk) * GSTR + kk], hb);
#pragma unroll
    for (int mi = 0; mi < 2; ++mi)
#pragma unroll
      for (int ni = 0; ni < 4; ++ni)
        acc[mi][ni] = wmma16(a[mi], bf[ni], acc[mi][ni]);
  }

#pragma unroll
  for (int mi = 0; mi < 2; ++mi)
#pragma unroll
    for (int ni = 0; ni < 4; ++ni)
#pragma unroll
      for (int r = 0; r < 8; ++r) {
        int mg = mb * 128 + Mb + mi * 16 + r + hb * 8;  // token row
        int nc = Nb + ni * 16 + lnk;                    // 0..127 within which
        float v = acc[mi][ni][r] + qkv_b[nb * 128 + nc];
        int hh = nc >> 5, d = nc & 31;
        int bI = mg >> 8, li = mg & 255;
        qkv16[((((size_t)nb * BW + bI) * HEADS + hh) << 13) + (li << 5) + d] =
            (f16)v;
      }
}

// ---------------------------------------------------------------------------
// Kernel 3: fused flash-style window attention, one block per (window, head).
// q/k staged [token][d] (K-contiguous for both A and B fragments),
// v staged transposed [d][token] so PV B-fragments are contiguous.
// Online softmax over 8 k-steps of 32 tokens.
// ---------------------------------------------------------------------------
__global__ __launch_bounds__(256) void attn_kernel(
    const f16* __restrict__ qkv16, const float* __restrict__ mask,
    const float* __restrict__ tau, const float* __restrict__ biasws,
    f16* __restrict__ o16) {
  const int b = blockIdx.x >> 2, h = blockIdx.x & 3;
  __shared__ __align__(16) f16 qs[L_TOK * QSTR];     // 20 KB
  __shared__ __align__(16) f16 ks[L_TOK * QSTR];     // 20 KB
  __shared__ __align__(16) f16 vst[HD * VSTR];       // 16.5 KB, transposed
  __shared__ __align__(16) f16 pscr[8 * 32 * QSTR];  // 20 KB
  __shared__ float qn[256], kn[256];
  const int tid = threadIdx.x;

  const size_t base = ((size_t)b * HEADS + h) << 13;  // *8192 halfs
  const size_t wstride = (size_t)BW * HEADS * 8192;   // q/k/v plane stride
#pragma unroll
  for (int c = 0; c < 4; ++c) {
    int g = c * 256 + tid;          // 1024 granules of 8 halfs
    int token = g >> 2, d8 = (g & 3) << 3;
    size_t off = (size_t)g * 8;
    *(uint4*)&qs[token * QSTR + d8] = *(const uint4*)&qkv16[base + off];
    *(uint4*)&ks[token * QSTR + d8] = *(const uint4*)&qkv16[wstride + base + off];
    uint4 vr = *(const uint4*)&qkv16[2 * wstride + base + off];
    const f16* vsrc = (const f16*)&vr;
#pragma unroll
    for (int i = 0; i < 8; ++i) vst[(d8 + i) * VSTR + token] = vsrc[i];
  }
  __syncthreads();
  {
    int row = tid;
    float sq = 0.f, sk = 0.f;
#pragma unroll
    for (int c4 = 0; c4 < 4; ++c4) {
      v8h q8 = *(const v8h*)&qs[row * QSTR + c4 * 8];
      v8h k8 = *(const v8h*)&ks[row * QSTR + c4 * 8];
#pragma unroll
      for (int i = 0; i < 8; ++i) {
        float qv = (float)q8[i]; sq += qv * qv;
        float kv = (float)k8[i]; sk += kv * kv;
      }
    }
    qn[row] = sqrtf(sq);
    kn[row] = sqrtf(sk);
  }
  __syncthreads();

  const int lane = tid & 31, wid = tid >> 5;
  const int lnk = lane & 15, hb = lane >> 4;
  const int Mb = wid * 32;
  const float tinv = 1.0f / fmaxf(tau[h], 0.01f);
  const float* maskp = mask + (size_t)(b & (NWIN - 1)) * 65536;
  const float* biasp = biasws + (size_t)h * 65536;

  v16h aq[2];
#pragma unroll
  for (int mi = 0; mi < 2; ++mi)
    aq[mi] = fragA(&qs[(Mb + mi * 16 + lnk) * QSTR], hb);

  v8f o[2][2];
  float mrun[2][8], lrun[2][8];
#pragma unroll
  for (int mi = 0; mi < 2; ++mi) {
#pragma unroll
    for (int no = 0; no < 2; ++no) o[mi][no] = zero8();
#pragma unroll
    for (int r = 0; r < 8; ++r) { mrun[mi][r] = -1e30f; lrun[mi][r] = 0.f; }
  }

  for (int kt = 0; kt < 8; ++kt) {
    // ---- S = q @ k^T over this 32-token slab ----
    v16h bk[2];
#pragma unroll
    for (int ni = 0; ni < 2; ++ni)
      bk[ni] = fragB(&ks[(kt * 32 + ni * 16 + lnk) * QSTR], hb);
    v8f c[2][2];
#pragma unroll
    for (int mi = 0; mi < 2; ++mi)
#pragma unroll
      for (int ni = 0; ni < 2; ++ni)
        c[mi][ni] = wmma16(aq[mi], bk[ni], zero8());

    // ---- cosine scale + bias + mask + online softmax ----
#pragma unroll
    for (int mi = 0; mi < 2; ++mi)
#pragma unroll
      for (int r = 0; r < 8; ++r) {
        int row = Mb + mi * 16 + r + hb * 8;
        float qnr = qn[row];
        float sv0, sv1;
        {
          int col = kt * 32 + lnk;
          float den = fmaxf(qnr * kn[col], 1e-6f);
          sv0 = c[mi][0][r] / den * tinv + biasp[row * 256 + col] +
                maskp[row * 256 + col];
          col += 16;
          den = fmaxf(qnr * kn[col], 1e-6f);
          sv1 = c[mi][1][r] / den * tinv + biasp[row * 256 + col] +
                maskp[row * 256 + col];
        }
        float mx = fmaxf(sv0, sv1);
        mx = fmaxf(mx, __shfl_xor(mx, 1));
        mx = fmaxf(mx, __shfl_xor(mx, 2));
        mx = fmaxf(mx, __shfl_xor(mx, 4));
        mx = fmaxf(mx, __shfl_xor(mx, 8));
        float mnew = fmaxf(mrun[mi][r], mx);
        float corr = __expf(mrun[mi][r] - mnew);
        float p0 = __expf(sv0 - mnew), p1 = __expf(sv1 - mnew);
        float rs = p0 + p1;
        rs += __shfl_xor(rs, 1);
        rs += __shfl_xor(rs, 2);
        rs += __shfl_xor(rs, 4);
        rs += __shfl_xor(rs, 8);
        lrun[mi][r] = lrun[mi][r] * corr + rs;
        mrun[mi][r] = mnew;
        o[mi][0][r] *= corr;
        o[mi][1][r] *= corr;
        int pr = mi * 16 + r + hb * 8;
        pscr[(wid * 32 + pr) * QSTR + lnk] = (f16)p0;
        pscr[(wid * 32 + pr) * QSTR + 16 + lnk] = (f16)p1;
      }

    // ---- O += P @ V ----
    v16h ap[2];
#pragma unroll
    for (int mi = 0; mi < 2; ++mi)
      ap[mi] = fragA(&pscr[(wid * 32 + mi * 16 + lnk) * QSTR], hb);
    v16h bv[2];
#pragma unroll
    for (int no = 0; no < 2; ++no) {
      const f16* pv = &vst[(no * 16 + lnk) * VSTR + kt * 32];
      bv[no] = fragB(pv, hb);
    }
#pragma unroll
    for (int mi = 0; mi < 2; ++mi)
#pragma unroll
      for (int no = 0; no < 2; ++no)
        o[mi][no] = wmma16(ap[mi], bv[no], o[mi][no]);
  }

  // ---- normalize and write (B, L, C) f16 ----
#pragma unroll
  for (int mi = 0; mi < 2; ++mi)
#pragma unroll
    for (int no = 0; no < 2; ++no)
#pragma unroll
      for (int r = 0; r < 8; ++r) {
        int row = Mb + mi * 16 + r + hb * 8;
        float val = o[mi][no][r] / lrun[mi][r];
        int d = no * 16 + lnk;
        o16[((size_t)b * L_TOK + row) * DIM + h * HD + d] = (f16)val;
      }
}

// ---------------------------------------------------------------------------
// Kernel 4: output projection. out(262144 x 128) = o16 @ proj_w^T + b  (f32)
// ---------------------------------------------------------------------------
__global__ __launch_bounds__(256) void proj_gemm_kernel(
    const f16* __restrict__ a16, const f16* __restrict__ w16,
    const float* __restrict__ pb, float* __restrict__ out) {
  const int mb = blockIdx.x;
  __shared__ __align__(16) f16 As[128 * GSTR];
  __shared__ __align__(16) f16 Bs[128 * GSTR];
  const int tid = threadIdx.x;

#pragma unroll
  for (int c = 0; c < 8; ++c) {
    int g = c * 256 + tid;
    int row = g >> 4, k8 = (g & 15) << 3;
    *(uint4*)&As[row * GSTR + k8] =
        *(const uint4*)&a16[(size_t)(mb * 128 + row) * 128 + k8];
  }
#pragma unroll
  for (int c = 0; c < 8; ++c) {
    int g = c * 256 + tid;
    int n = g >> 4, k8 = (g & 15) << 3;
    *(uint4*)&Bs[n * GSTR + k8] = *(const uint4*)&w16[(size_t)n * 128 + k8];
  }
  __syncthreads();

  const int lane = tid & 31, wid = tid >> 5;
  const int lnk = lane & 15, hb = lane >> 4;
  const int Mb = (wid & 3) * 32, Nb = (wid >> 2) * 64;

  v8f acc[2][4];
#pragma unroll
  for (int mi = 0; mi < 2; ++mi)
#pragma unroll
    for (int ni = 0; ni < 4; ++ni) acc[mi][ni] = zero8();

#pragma unroll
  for (int kk = 0; kk < 128; kk += 32) {
    v16h a[2];
#pragma unroll
    for (int mi = 0; mi < 2; ++mi)
      a[mi] = fragA(&As[(Mb + mi * 16 + lnk) * GSTR + kk], hb);
    v16h bf[4];
#pragma unroll
    for (int ni = 0; ni < 4; ++ni)
      bf[ni] = fragB(&Bs[(Nb + ni * 16 + lnk) * GSTR + kk], hb);
#pragma unroll
    for (int mi = 0; mi < 2; ++mi)
#pragma unroll
      for (int ni = 0; ni < 4; ++ni)
        acc[mi][ni] = wmma16(a[mi], bf[ni], acc[mi][ni]);
  }

#pragma unroll
  for (int mi = 0; mi < 2; ++mi)
#pragma unroll
    for (int ni = 0; ni < 4; ++ni)
#pragma unroll
      for (int r = 0; r < 8; ++r) {
        int mg = mb * 128 + Mb + mi * 16 + r + hb * 8;
        int ng = Nb + ni * 16 + lnk;
        out[(size_t)mg * DIM + ng] = acc[mi][ni][r] + pb[ng];
      }
}

// ---------------------------------------------------------------------------
extern "C" void kernel_launch(void* const* d_in, const int* in_sizes, int n_in,
                              void* d_out, int out_size, void* d_ws,
                              size_t ws_size, hipStream_t stream) {
  const float* x      = (const float*)d_in[0];
  const float* mask   = (const float*)d_in[1];
  const float* qkv_w  = (const float*)d_in[2];
  const float* qkv_b  = (const float*)d_in[3];
  const float* proj_w = (const float*)d_in[4];
  const float* proj_b = (const float*)d_in[5];
  const float* tau    = (const float*)d_in[6];
  const float* mlp_w1 = (const float*)d_in[7];
  const float* mlp_b1 = (const float*)d_in[8];
  const float* mlp_w2 = (const float*)d_in[9];
  const float* mlp_b2 = (const float*)d_in[10];
  float* out = (float*)d_out;

  char* wsb = (char*)d_ws;
  f16*   qkvw16  = (f16*)(wsb);                           //  98,304 B
  f16*   projw16 = (f16*)(wsb + 98304);                   //  32,768 B
  float* biasws  = (float*)(wsb + 131072);                //   1 MiB
  f16*   qkv16   = (f16*)(wsb + 1179648);                 // 192 MiB
  f16*   o16     = (f16*)(wsb + 1179648 + 201326592ULL);  //  64 MiB

  convert_w_kernel<<<192, 256, 0, stream>>>(qkv_w, proj_w, qkvw16, projw16);
  bias_kernel<<<256, 256, 0, stream>>>(mlp_w1, mlp_b1, mlp_w2, mlp_b2, biasws);
  qkv_gemm_kernel<<<dim3(2048, 3), 256, 0, stream>>>(x, qkvw16, qkv_b, qkv16);
  attn_kernel<<<BW * HEADS, 256, 0, stream>>>(qkv16, mask, tau, biasws, o16);
  proj_gemm_kernel<<<dim3(2048, 1), 256, 0, stream>>>(o16, projw16, proj_b,
                                                      out);
}
